// xDeepFM_47528108098261
// MI455X (gfx1250) — matrix operand order repaired and checked by reading, not compiled, not run on hardware
//
#include <hip/hip_runtime.h>
#include <cstdint>
#include <cstddef>

typedef __attribute__((ext_vector_type(16))) __bf16 v16bf;
typedef __attribute__((ext_vector_type(8)))  float  v8f;
typedef unsigned int uint32;

// ---------------- constants ----------------
#define BATCH   4096
#define NSF     26
#define DNSF    13
#define DIM     16
#define VOC     100000
#define NFLD    39          // NS + DNS
#define FCIN    128
#define K0PAD   448         // din=429 padded to 14*32
#define D1      1024
#define D2      512
#define D3      256
#define KT_W1   14
#define KT_W2   32
#define KT_W3   16
#define KT_C0   48          // ceil(39*39/32)
#define KT_C12  156         // 128*39/32

// ---------------- helpers ----------------
__device__ __forceinline__ uint32 f2bf(float x) {
    uint32 b = __float_as_uint(x);
    return (b + 0x7fffu + ((b >> 16) & 1u)) >> 16;   // RNE f32->bf16
}
__device__ __forceinline__ uint32 pack2(float lo, float hi) {
    return f2bf(lo) | (f2bf(hi) << 16);
}
__device__ __forceinline__ float bf2f(uint16_t h) {
    return __uint_as_float(((uint32)h) << 16);
}

union FragU { v16bf v; uint32 u[8]; };

// A fragment (16x32 bf16) from LDS tile stored as [16 rows][17 dword-pairs]
// layout: lane half h, VGPR j holds K = (j>=4?16:0) + h*8 + (j&3)*2 (+1)
__device__ __forceinline__ v16bf load_a_frag(const uint32* a_lds, int lane) {
    FragU r;
    const int row = lane & 15, half = lane >> 4;
#pragma unroll
    for (int j = 0; j < 8; ++j) {
        int pair = (j >> 2) * 8 + half * 4 + (j & 3);
        r.u[j] = a_lds[row * 17 + pair];
    }
    return r.v;
}

// B fragment (32x16 bf16) pre-packed in global: 8 contiguous dwords per lane
__device__ __forceinline__ v16bf load_b_frag(const uint32* p) {
    FragU r;
    const uint4* q = (const uint4*)p;
    uint4 a = q[0], b = q[1];
    r.u[0]=a.x; r.u[1]=a.y; r.u[2]=a.z; r.u[3]=a.w;
    r.u[4]=b.x; r.u[5]=b.y; r.u[6]=b.z; r.u[7]=b.w;
    return r.v;
}

// async copy of one dword global -> LDS (gfx1250 ASYNCcnt path)
__device__ __forceinline__ void async_copy_b32(uint32* lds_dst, const uint32* gsrc) {
    uint32 ldsa = (uint32)(uintptr_t)lds_dst;           // low 32 bits = LDS offset
    unsigned long long ga = (unsigned long long)(uintptr_t)gsrc;
    asm volatile("global_load_async_to_lds_b32 %0, %1, off"
                 :: "v"(ldsa), "v"(ga) : "memory");
}
__device__ __forceinline__ void wait_async0() {
    asm volatile("s_wait_asynccnt 0" ::: "memory");
}

// ---------------- weight fragment packers ----------------
// src W row-major [Ksrc, Nn]; dst fragments [Ktiles][Nn/16][32 lanes][8 dwords]
__global__ void pack_rowmajor(const float* __restrict__ W, uint32* __restrict__ dst,
                              int Ksrc, int Nn, int Ntiles) {
    int g = blockIdx.x * 256 + threadIdx.x;
    int j = g & 7, L = (g >> 3) & 31;
    int rest = g >> 8;
    int tn = rest % Ntiles, kt = rest / Ntiles;
    int col = tn * 16 + (L & 15);
    int Kb  = kt * 32 + (j >> 2) * 16 + (L >> 4) * 8 + (j & 3) * 2;
    float v0 = (Kb     < Ksrc) ? W[(size_t)Kb       * Nn + col] : 0.f;
    float v1 = (Kb + 1 < Ksrc) ? W[(size_t)(Kb + 1) * Nn + col] : 0.f;
    dst[g] = pack2(v0, v1);
}

// src W [F=128, C, 39]; effective B matrix is [CN, 128] with row K=(c*39+n), col f
__global__ void pack_cin(const float* __restrict__ W, uint32* __restrict__ dst, int C) {
    int g = blockIdx.x * 256 + threadIdx.x;
    int j = g & 7, L = (g >> 3) & 31;
    int rest = g >> 8;
    int tn = rest & 7, kt = rest >> 3;
    int f  = tn * 16 + (L & 15);
    int Kb = kt * 32 + (j >> 2) * 16 + (L >> 4) * 8 + (j & 3) * 2;
    int CN = C * 39;
    float v0 = 0.f, v1 = 0.f;
    if (Kb     < CN) v0 = W[((size_t)f * C + (Kb)     / 39) * 39 + (Kb)     % 39];
    if (Kb + 1 < CN) v1 = W[((size_t)f * C + (Kb + 1) / 39) * 39 + (Kb + 1) % 39];
    dst[g] = pack2(v0, v1);
}

// ---------------- gather: embeddings, h0, x0, linear logits ----------------
__global__ __launch_bounds__(256)
void gather_embed(const float* __restrict__ dense_x, const int* __restrict__ disc,
                  const float* __restrict__ lin_emb, const float* __restrict__ emb_W,
                  const float* __restrict__ dense_W, const float* __restrict__ dense_b,
                  const float* __restrict__ dense_proj,
                  uint16_t* __restrict__ h0, float* __restrict__ x0,
                  float* __restrict__ lin) {
    const int b = blockIdx.x, tid = threadIdx.x;
    __shared__ int   idxs[NSF];
    __shared__ float red[64];
    if (tid < NSF) idxs[tid] = disc[b * NSF + tid];
    __syncthreads();
    for (int t = tid; t < NSF * DIM; t += 256) {
        int f = t >> 4, d = t & 15;
        float v = emb_W[((size_t)f * VOC + idxs[f]) * DIM + d];
        h0[(size_t)b * K0PAD + t] = (uint16_t)f2bf(v);
        x0[((size_t)b * NFLD + f) * DIM + d] = v;
    }
    if (tid < DNSF)
        h0[(size_t)b * K0PAD + NSF * DIM + tid] = (uint16_t)f2bf(dense_x[b * DNSF + tid]);
    if (tid < (K0PAD - NSF * DIM - DNSF))
        h0[(size_t)b * K0PAD + NSF * DIM + DNSF + tid] = 0;
    for (int t = tid; t < DNSF * DIM; t += 256) {
        int j = t >> 4, d = t & 15;
        x0[((size_t)b * NFLD + NSF + j) * DIM + d] =
            dense_x[b * DNSF + j] * dense_proj[j * DIM + d];
    }
    float term = 0.f;
    if (tid < NSF)        term = lin_emb[(size_t)tid * VOC + idxs[tid]];
    else if (tid < NFLD)  term = dense_x[b * DNSF + (tid - NSF)] * dense_W[tid - NSF];
    if (tid < 64) red[tid] = (tid < NFLD) ? term : 0.f;
    __syncthreads();
    if (tid == 0) {
        float s = 0.f;
        for (int i = 0; i < NFLD; ++i) s += red[i];
        lin[b] = s + dense_b[0];
    }
}

// ---------------- bf16 WMMA GEMM + bias + BN(eval) + ReLU ----------------
// A [M, Kpad] bf16 row-major ; Bf fragments [Kpad/32][Nout/16][32][8] ; Out [M, Nout] bf16
__global__ __launch_bounds__(256)
void gemm_bn_relu(const uint16_t* __restrict__ A, const uint32* __restrict__ Bf,
                  const float* __restrict__ bias, const float* __restrict__ gamma,
                  const float* __restrict__ beta, uint16_t* __restrict__ Out,
                  int Kpad, int Nout) {
    const int tid = threadIdx.x, lane = tid & 31, w = tid >> 5;
    const int m0 = blockIdx.x * 64;
    const int NtilesTot = Nout >> 4;
    const int msub = w & 3, ngrp = w >> 2;
    __shared__ uint32 a_lds[64 * 17];
    v8f acc[4];
#pragma unroll
    for (int t = 0; t < 4; ++t) acc[t] = (v8f){0.f,0.f,0.f,0.f,0.f,0.f,0.f,0.f};

    const uint32* Asrc = (const uint32*)A;
    const int Ksteps = Kpad >> 5;
    for (int kt = 0; kt < Ksteps; ++kt) {
        // async global->LDS staging of the 64x32 bf16 A tile (1024 dwords)
#pragma unroll
        for (int it = 0; it < 4; ++it) {
            int idx = tid + it * 256;
            int row = idx >> 4, pr = idx & 15;
            async_copy_b32(&a_lds[row * 17 + pr],
                           Asrc + (size_t)(m0 + row) * (Kpad >> 1) + kt * 16 + pr);
        }
        wait_async0();
        __syncthreads();
        v16bf a = load_a_frag(&a_lds[msub * 16 * 17], lane);
#pragma unroll
        for (int t = 0; t < 4; ++t) {
            int tn = blockIdx.y * 8 + ngrp * 4 + t;
            v16bf b = load_b_frag(Bf + ((size_t)(kt * NtilesTot + tn) * 32 + lane) * 8);
            // speculative WGP-scope prefetch of next K-step's fragment stream
            __builtin_prefetch(Bf + ((size_t)((kt + 1) * NtilesTot + tn) * 32 + lane) * 8, 0, 3);
            acc[t] = __builtin_amdgcn_wmma_f32_16x16x32_bf16(
                         false, a, false, b, (short)0, acc[t], false, false);
        }
        __syncthreads();
    }
    const float rs = rsqrtf(1.0f + 1e-5f);
    const int half = lane >> 4;
#pragma unroll
    for (int t = 0; t < 4; ++t) {
        int col = (blockIdx.y * 8 + ngrp * 4 + t) * 16 + (lane & 15);
        float bi = bias[col], sc = gamma[col] * rs, bb = beta[col];
#pragma unroll
        for (int r = 0; r < 8; ++r) {
            int row = m0 + msub * 16 + half * 8 + r;
            float v = (acc[t][r] + bi) * sc + bb;
            v = fmaxf(v, 0.f);
            Out[(size_t)row * Nout + col] = (uint16_t)f2bf(v);
        }
    }
}

// ---------------- fused CIN layer ----------------
// out[b,f,d] = sum_{c,n} xj[b,c,d]*x0[b,n,d]*W[f,c,n]  == T(32xCN) @ Wt(CNx128) per (b32,d)
// 32 batch rows (two 16-row M tiles) per block: 2 WMMAs per B-fragment load.
__global__ __launch_bounds__(256)
void cin_layer(const float* __restrict__ xj, const float* __restrict__ x0,
               const uint32* __restrict__ Wf, float* __restrict__ out,
               int C, int Ktiles) {
    const int tid = threadIdx.x, lane = tid & 31, w = tid >> 5;
    const int b0 = blockIdx.x * 32, d = blockIdx.y;
    __shared__ float  xjt[32 * FCIN];
    __shared__ float  x0t[32 * NFLD];
    __shared__ uint32 a_lds[32 * 17];

    for (int i = tid; i < 32 * C; i += 256) {
        int m = i / C, c = i % C;
        xjt[i] = xj[((size_t)(b0 + m) * C + c) * DIM + d];
    }
    for (int i = tid; i < 32 * NFLD; i += 256) {
        int m = i / NFLD, n = i % NFLD;
        x0t[i] = x0[((size_t)(b0 + m) * NFLD + n) * DIM + d];
    }
    __syncthreads();

    v8f acc0 = (v8f){0.f,0.f,0.f,0.f,0.f,0.f,0.f,0.f};
    v8f acc1 = (v8f){0.f,0.f,0.f,0.f,0.f,0.f,0.f,0.f};
    const int pr = tid & 15, mlo = tid >> 4;     // thread builds rows mlo and mlo+16
    // incremental (c,n) tracking for k0 = kt*32 + 2*pr and k1 = k0+1  (32 < 39)
    int c0 = 0, n0 = 2 * pr;
    int c1 = 0, n1 = 2 * pr + 1;
    for (int kt = 0; kt < Ktiles; ++kt) {
        float a00 = 0.f, a01 = 0.f, a10 = 0.f, a11 = 0.f;
        if (c0 < C) {
            a00 = xjt[mlo * C + c0]        * x0t[mlo * NFLD + n0];
            a10 = xjt[(mlo + 16) * C + c0] * x0t[(mlo + 16) * NFLD + n0];
        }
        if (c1 < C) {
            a01 = xjt[mlo * C + c1]        * x0t[mlo * NFLD + n1];
            a11 = xjt[(mlo + 16) * C + c1] * x0t[(mlo + 16) * NFLD + n1];
        }
        a_lds[mlo * 17 + pr]        = pack2(a00, a01);
        a_lds[(mlo + 16) * 17 + pr] = pack2(a10, a11);
        n0 += 32; if (n0 >= NFLD) { n0 -= NFLD; ++c0; }
        n1 += 32; if (n1 >= NFLD) { n1 -= NFLD; ++c1; }
        __syncthreads();
        v16bf aF0 = load_a_frag(a_lds, lane);
        v16bf aF1 = load_a_frag(a_lds + 16 * 17, lane);
        v16bf b   = load_b_frag(Wf + ((size_t)(kt * 8 + w) * 32 + lane) * 8);
        // speculative WGP-scope prefetch of next K-step's fragment stream
        __builtin_prefetch(Wf + ((size_t)((kt + 1) * 8 + w) * 32 + lane) * 8, 0, 3);
        acc0 = __builtin_amdgcn_wmma_f32_16x16x32_bf16(
                   false, aF0, false, b, (short)0, acc0, false, false);
        acc1 = __builtin_amdgcn_wmma_f32_16x16x32_bf16(
                   false, aF1, false, b, (short)0, acc1, false, false);
        __syncthreads();
    }
    const int half = lane >> 4;
    const int f = w * 16 + (lane & 15);
#pragma unroll
    for (int r = 0; r < 8; ++r) {
        int bb0 = b0 + half * 8 + r;
        out[((size_t)bb0 * FCIN + f) * DIM + d] = acc0[r];
        int bb1 = b0 + 16 + half * 8 + r;
        out[((size_t)bb1 * FCIN + f) * DIM + d] = acc1[r];
    }
}

// ---------------- p[b,f] = sum_d out[b,f,d] ----------------
__global__ void reduce_pk(const float* __restrict__ cino, float* __restrict__ p, int off) {
    int g = blockIdx.x * 256 + threadIdx.x;      // over B*128
    int b = g >> 7, f = g & 127;
    const float4* src = (const float4*)(cino + ((size_t)b * FCIN + f) * DIM);
    float s = 0.f;
#pragma unroll
    for (int i = 0; i < 4; ++i) { float4 v = src[i]; s += v.x + v.y + v.z + v.w; }
    p[(size_t)b * (3 * FCIN) + off + f] = s;
}

// ---------------- finalize: lin + dnn + cin logits ----------------
__global__ __launch_bounds__(128)
void finalize(const float* __restrict__ lin, const uint16_t* __restrict__ h3,
              const float* __restrict__ Wout, const float* __restrict__ bout,
              const float* __restrict__ p, const float* __restrict__ cW,
              const float* __restrict__ cb, float* __restrict__ out) {
    const int b = blockIdx.x, t = threadIdx.x;
    __shared__ float red[128];
    float s = 0.f;
    for (int j = t; j < D3; j += 128)        s += bf2f(h3[(size_t)b * D3 + j]) * Wout[j];
    for (int j = t; j < 3 * FCIN; j += 128)  s += p[(size_t)b * (3 * FCIN) + j] * cW[j];
    red[t] = s;
    __syncthreads();
    for (int st = 64; st > 0; st >>= 1) { if (t < st) red[t] += red[t + st]; __syncthreads(); }
    if (t == 0) out[b] = red[0] + lin[b] + bout[0] + cb[0];
}

// ---------------- host launch ----------------
extern "C" void kernel_launch(void* const* d_in, const int* in_sizes, int n_in,
                              void* d_out, int out_size, void* d_ws, size_t ws_size,
                              hipStream_t stream) {
    const float* dense_x    = (const float*)d_in[0];
    const int*   discrete_x = (const int*)  d_in[1];
    const float* lin_emb    = (const float*)d_in[2];
    const float* emb_W      = (const float*)d_in[3];
    const float* dense_W    = (const float*)d_in[4];
    const float* dense_b    = (const float*)d_in[5];
    const float* W1         = (const float*)d_in[6];
    const float* b1         = (const float*)d_in[7];
    const float* bn1_g      = (const float*)d_in[8];
    const float* bn1_b      = (const float*)d_in[9];
    const float* W2         = (const float*)d_in[10];
    const float* b2         = (const float*)d_in[11];
    const float* bn2_g      = (const float*)d_in[12];
    const float* bn2_b      = (const float*)d_in[13];
    const float* W3         = (const float*)d_in[14];
    const float* b3         = (const float*)d_in[15];
    const float* bn3_g      = (const float*)d_in[16];
    const float* bn3_b      = (const float*)d_in[17];
    const float* Wout       = (const float*)d_in[18];
    const float* bout       = (const float*)d_in[19];
    const float* dense_proj = (const float*)d_in[20];
    const float* cin_W0     = (const float*)d_in[21];
    const float* cin_W1     = (const float*)d_in[22];
    const float* cin_W2     = (const float*)d_in[23];
    const float* cin_out_W  = (const float*)d_in[24];
    const float* cin_out_b  = (const float*)d_in[25];
    float* out = (float*)d_out;

    // workspace carve-up (256B aligned)
    char* ws = (char*)d_ws;
    size_t off = 0;
    auto carve = [&](size_t bytes) { char* r = ws + off; off = (off + bytes + 255) & ~(size_t)255; return r; };
    uint16_t* h0  = (uint16_t*)carve((size_t)BATCH * K0PAD * 2);
    uint16_t* h1  = (uint16_t*)carve((size_t)BATCH * D1 * 2);
    uint16_t* h2  = (uint16_t*)carve((size_t)BATCH * D2 * 2);
    uint16_t* h3  = (uint16_t*)carve((size_t)BATCH * D3 * 2);
    float*    x0  = (float*)   carve((size_t)BATCH * NFLD * DIM * 4);
    float*    xjA = (float*)   carve((size_t)BATCH * FCIN * DIM * 4);
    float*    xjB = (float*)   carve((size_t)BATCH * FCIN * DIM * 4);
    float*    pbf = (float*)   carve((size_t)BATCH * 3 * FCIN * 4);
    float*    lin = (float*)   carve((size_t)BATCH * 4);
    uint32*   w1f = (uint32*)  carve((size_t)KT_W1 * (D1/16) * 256 * 4);
    uint32*   w2f = (uint32*)  carve((size_t)KT_W2 * (D2/16) * 256 * 4);
    uint32*   w3f = (uint32*)  carve((size_t)KT_W3 * (D3/16) * 256 * 4);
    uint32*   c0f = (uint32*)  carve((size_t)KT_C0  * 8 * 256 * 4);
    uint32*   c1f = (uint32*)  carve((size_t)KT_C12 * 8 * 256 * 4);
    uint32*   c2f = (uint32*)  carve((size_t)KT_C12 * 8 * 256 * 4);
    (void)ws_size; (void)in_sizes; (void)n_in; (void)out_size;

    // 1. pack weights into WMMA B-fragment layout
    pack_rowmajor<<<KT_W1 * (D1/16), 256, 0, stream>>>(W1, w1f, 429,  D1, D1/16);
    pack_rowmajor<<<KT_W2 * (D2/16), 256, 0, stream>>>(W2, w2f, D1,   D2, D2/16);
    pack_rowmajor<<<KT_W3 * (D3/16), 256, 0, stream>>>(W3, w3f, D2,   D3, D3/16);
    pack_cin<<<KT_C0  * 8, 256, 0, stream>>>(cin_W0, c0f, NFLD);
    pack_cin<<<KT_C12 * 8, 256, 0, stream>>>(cin_W1, c1f, FCIN);
    pack_cin<<<KT_C12 * 8, 256, 0, stream>>>(cin_W2, c2f, FCIN);

    // 2. gather embeddings + linear part
    gather_embed<<<BATCH, 256, 0, stream>>>(dense_x, discrete_x, lin_emb, emb_W,
                                            dense_W, dense_b, dense_proj, h0, x0, lin);

    // 3. DNN tower (bf16 WMMA GEMMs, fused bias+BN+ReLU)
    gemm_bn_relu<<<dim3(BATCH/64, D1/128), 256, 0, stream>>>(h0, w1f, b1, bn1_g, bn1_b, h1, K0PAD, D1);
    gemm_bn_relu<<<dim3(BATCH/64, D2/128), 256, 0, stream>>>(h1, w2f, b2, bn2_g, bn2_b, h2, D1, D2);
    gemm_bn_relu<<<dim3(BATCH/64, D3/128), 256, 0, stream>>>(h2, w3f, b3, bn3_g, bn3_b, h3, D2, D3);

    // 4. CIN tower (fused outer-product WMMA layers, 32 rows/block)
    cin_layer<<<dim3(BATCH/32, DIM), 256, 0, stream>>>(x0,  x0, c0f, xjA, NFLD, KT_C0);
    reduce_pk<<<BATCH * FCIN / 256, 256, 0, stream>>>(xjA, pbf, 0);
    cin_layer<<<dim3(BATCH/32, DIM), 256, 0, stream>>>(xjA, x0, c1f, xjB, FCIN, KT_C12);
    reduce_pk<<<BATCH * FCIN / 256, 256, 0, stream>>>(xjB, pbf, FCIN);
    cin_layer<<<dim3(BATCH/32, DIM), 256, 0, stream>>>(xjB, x0, c2f, xjA, FCIN, KT_C12);
    reduce_pk<<<BATCH * FCIN / 256, 256, 0, stream>>>(xjA, pbf, 2 * FCIN);

    // 5. combine logits
    finalize<<<BATCH, 128, 0, stream>>>(lin, h3, Wout, bout, pbf, cin_out_W, cin_out_b, out);
}
